// MultiHeadAttention_88854283420200
// MI455X (gfx1250) — matrix operand order repaired and checked by reading, not compiled, not run on hardware
//
#include <hip/hip_runtime.h>

// ---------------------------------------------------------------------------
// MHA forward for MI455X (gfx1250, wave32, WMMA bf16 -> f32 accum)
// B=2, S=2048, H=1024, nh=16, hd=64
// ---------------------------------------------------------------------------

typedef __attribute__((ext_vector_type(16))) __bf16 v16bf;
typedef __attribute__((ext_vector_type(8)))  float  v8f;

#define NB      2
#define SEQ     2048
#define HID     1024
#define NH      16
#define HD      64
#define NROWS   (NB * SEQ)          // 4096

#define LDA 40                      // LDS stride (halfwords) for 32-wide tiles
#define LDB 72                      // LDS stride (halfwords) for 64-wide tiles

static __device__ __forceinline__ unsigned short bf16_rne(float f) {
    unsigned u = __float_as_uint(f);
    unsigned r = u + 0x7FFFu + ((u >> 16) & 1u);
    return (unsigned short)(r >> 16);
}

// Pack two f32 into one dword of bf16x2. Prefer the single-VALU-op
// v_cvt_pk_bf16_f32 when the toolchain exposes it.
static __device__ __forceinline__ unsigned pack2_bf16(float lo, float hi) {
#if __has_builtin(__builtin_amdgcn_cvt_pk_bf16_f32)
    typedef __attribute__((ext_vector_type(2))) __bf16 v2bf;
    v2bf r = __builtin_amdgcn_cvt_pk_bf16_f32(lo, hi);
    unsigned u;
    __builtin_memcpy(&u, &r, 4);
    return u;
#else
    return (unsigned)bf16_rne(lo) | ((unsigned)bf16_rne(hi) << 16);
#endif
}

union FragU { uint4 u[2]; v16bf v; };

// A-operand (16x32, MxK): lane = row M (0..15); half picks K groups
//   lanes 0-15 : K = 0..7, 16..23   lanes 16-31 : K = 8..15, 24..31
static __device__ __forceinline__ v16bf load_fragA(const unsigned short* tile, int ld) {
    const int lane = threadIdx.x & 31;
    const int r = lane & 15, hf = lane >> 4;
    const unsigned short* p = tile + r * ld + 8 * hf;
    FragU f;
    f.u[0] = *(const uint4*)(p);
    f.u[1] = *(const uint4*)(p + 16);
    return f.v;
}

// B-operand (32x16, KxN) sourced from B^T rows (row index N = lane&15):
//   lanes 0-15 : K = 0..7, 16..23   lanes 16-31 : K = 8..15, 24..31
static __device__ __forceinline__ v16bf load_fragB(const unsigned short* tileT, int ld) {
    const int lane = threadIdx.x & 31;
    const int n = lane & 15, hf = lane >> 4;
    const unsigned short* p = tileT + n * ld + 8 * hf;
    FragU f;
    f.u[0] = *(const uint4*)(p);
    f.u[1] = *(const uint4*)(p + 16);
    return f.v;
}

static __device__ __forceinline__ v8f wmma_bf16(v16bf a, v16bf b, v8f c) {
    return __builtin_amdgcn_wmma_f32_16x16x32_bf16(false, a, false, b, (short)0, c, false, false);
}

// ---------------------------------------------------------------------------
// Kernel 1: fused Q/K/V projection.  out = (x @ W^T + b) [*1/8 for Q]
// written as bf16 in [B, nh, S, hd] layout.
// Block tile 128x64, 4 waves; each wave: 32 rows x 64 cols = 8 WMMA / K-step
// grid = (NROWS/128, HID/64, 3), block = 128
// ---------------------------------------------------------------------------
__global__ __launch_bounds__(128) void qkv_proj_kernel(
    const float* __restrict__ q, const float* __restrict__ k, const float* __restrict__ v,
    const float* __restrict__ Wq, const float* __restrict__ bq,
    const float* __restrict__ Wk, const float* __restrict__ bk,
    const float* __restrict__ Wv, const float* __restrict__ bv,
    unsigned short* __restrict__ Qp, unsigned short* __restrict__ Kp,
    unsigned short* __restrict__ Vp)
{
    __shared__ __align__(16) unsigned short sA[128 * LDA];
    __shared__ __align__(16) unsigned short sW[64 * LDA];

    const int tid  = threadIdx.x;
    const int wv   = tid >> 5;
    const int lane = tid & 31, hf = lane >> 4, nn = lane & 15;
    const int n0 = blockIdx.x * 128, o0 = blockIdx.y * 64;

    const float* x; const float* W; const float* bias;
    unsigned short* outp; float scale = 1.0f;
    if (blockIdx.z == 0)      { x = q; W = Wq; bias = bq; outp = Qp; scale = 0.125f; }
    else if (blockIdx.z == 1) { x = k; W = Wk; bias = bk; outp = Kp; }
    else                      { x = v; W = Wv; bias = bv; outp = Vp; }

    v8f acc[2][4] = {};

    for (int k0 = 0; k0 < HID; k0 += 32) {
        __syncthreads();
        // A tile: 128 rows x 32 cols fp32 -> bf16   (1024 float4 slots)
#pragma unroll
        for (int i = 0; i < 8; i++) {
            const int li = tid + 128 * i;
            const int row = li >> 3, c4 = li & 7;
            const float4 fa = *(const float4*)(x + (size_t)(n0 + row) * HID + k0 + 4 * c4);
            uint2 pa;
            pa.x = pack2_bf16(fa.x, fa.y);
            pa.y = pack2_bf16(fa.z, fa.w);
            *(uint2*)&sA[row * LDA + 4 * c4] = pa;
        }
        // W tile: 64 rows x 32 cols fp32 -> bf16
#pragma unroll
        for (int i = 0; i < 4; i++) {
            const int li = tid + 128 * i;
            const int row = li >> 3, c4 = li & 7;
            const float4 fw = *(const float4*)(W + (size_t)(o0 + row) * HID + k0 + 4 * c4);
            uint2 pw;
            pw.x = pack2_bf16(fw.x, fw.y);
            pw.y = pack2_bf16(fw.z, fw.w);
            *(uint2*)&sW[row * LDA + 4 * c4] = pw;
            if (k0 + 32 < HID)   // pull next K-slab toward L2 (global_prefetch_b8)
                __builtin_prefetch(W + (size_t)(o0 + row) * HID + k0 + 32 + 4 * c4, 0, 0);
        }
        __syncthreads();

        v16bf bfr[4];
#pragma unroll
        for (int t = 0; t < 4; t++)
            bfr[t] = load_fragB(sW + 16 * t * LDA, LDA);
        v16bf af0 = load_fragA(sA + (32 * wv) * LDA,      LDA);
        v16bf af1 = load_fragA(sA + (32 * wv + 16) * LDA, LDA);
#pragma unroll
        for (int t = 0; t < 4; t++) {
            acc[0][t] = wmma_bf16(af0, bfr[t], acc[0][t]);
            acc[1][t] = wmma_bf16(af1, bfr[t], acc[1][t]);
        }
    }

#pragma unroll
    for (int a = 0; a < 2; a++) {
#pragma unroll
        for (int t = 0; t < 4; t++) {
#pragma unroll
            for (int vr = 0; vr < 8; vr++) {
                const int n = n0 + 32 * wv + 16 * a + vr + 8 * hf;  // global row (b*S+s)
                const int o = o0 + 16 * t + nn;                     // global col (head*64+d)
                const float val = (acc[a][t][vr] + bias[o]) * scale;
                const int b = n >> 11, s = n & (SEQ - 1);
                const int head = o >> 6, d = o & (HD - 1);
                outp[((size_t)(b * NH + head) * SEQ + s) * HD + d] = bf16_rne(val);
            }
        }
    }
}

// ---------------------------------------------------------------------------
// Kernel 2: flash attention per (b,h).  Block = 64 query rows, 4 waves
// grid = (SEQ/64, NB*NH), block = 128
// ---------------------------------------------------------------------------
__global__ __launch_bounds__(128) void attn_kernel(
    const unsigned short* __restrict__ Qp, const unsigned short* __restrict__ Kp,
    const unsigned short* __restrict__ Vp, unsigned short* __restrict__ Obuf)
{
    __shared__ __align__(16) unsigned short sQ [64 * LDB];
    __shared__ __align__(16) unsigned short sK [64 * LDB];
    __shared__ __align__(16) unsigned short sVt[64 * LDB];      // V transposed [d][kv]
    __shared__ __align__(16) unsigned short sP [4 * 16 * LDB];  // per-wave P slab

    const int tid = threadIdx.x, wv = tid >> 5;
    const int lane = tid & 31, hf = lane >> 4, nn = lane & 15;
    const int bh = blockIdx.y;
    const int q0 = blockIdx.x * 64;

    const unsigned short* Qb = Qp + (size_t)bh * SEQ * HD;
    const unsigned short* Kb = Kp + (size_t)bh * SEQ * HD;
    const unsigned short* Vb = Vp + (size_t)bh * SEQ * HD;

    // ---- load Q tile (64 x 64 bf16), keep per-wave fragments in registers
#pragma unroll
    for (int i = 0; i < 4; i++) {
        const int li = tid + 128 * i;
        const int row = li >> 3, j = li & 7;
        *(uint4*)&sQ[row * LDB + 8 * j] =
            *(const uint4*)(Qb + (size_t)(q0 + row) * HD + 8 * j);
    }
    __syncthreads();
    v16bf qfrag[2];
    qfrag[0] = load_fragA(sQ + 16 * wv * LDB,      LDB);
    qfrag[1] = load_fragA(sQ + 16 * wv * LDB + 32, LDB);

    float mrow[8], lrow[8];
    v8f oacc[4] = {};
#pragma unroll
    for (int i = 0; i < 8; i++) { mrow[i] = -3.0e38f; lrow[i] = 0.0f; }

    unsigned short* sPw = sP + wv * 16 * LDB;

    for (int kv0 = 0; kv0 < SEQ; kv0 += 64) {
        __syncthreads();
        // ---- cooperative load: K straight, V transposed into LDS
#pragma unroll
        for (int i = 0; i < 4; i++) {
            const int li = tid + 128 * i;
            const int row = li >> 3, j = li & 7;
            *(uint4*)&sK[row * LDB + 8 * j] =
                *(const uint4*)(Kb + (size_t)(kv0 + row) * HD + 8 * j);
            const uint4 vvec = *(const uint4*)(Vb + (size_t)(kv0 + row) * HD + 8 * j);
            const unsigned short* e = (const unsigned short*)&vvec;
#pragma unroll
            for (int t = 0; t < 8; t++) sVt[(8 * j + t) * LDB + row] = e[t];
        }
        __syncthreads();

        // ---- scores = Qtile @ Ktile^T   (16 x 64 per wave, f32)
        v8f sc[4] = {};
        {
            v16bf kf0[4], kf1[4];
#pragma unroll
            for (int t = 0; t < 4; t++) {
                kf0[t] = load_fragB(sK + 16 * t * LDB,      LDB);
                kf1[t] = load_fragB(sK + 16 * t * LDB + 32, LDB);
            }
#pragma unroll
            for (int t = 0; t < 4; t++) {
                sc[t] = wmma_bf16(qfrag[0], kf0[t], sc[t]);
                sc[t] = wmma_bf16(qfrag[1], kf1[t], sc[t]);
            }
        }

        // ---- online softmax (row = vr + 8*hf; reduce across the 16 col-lanes)
        float newm[8], alpha[8], rsum[8];
#pragma unroll
        for (int vr = 0; vr < 8; vr++) {
            float tm = fmaxf(fmaxf(sc[0][vr], sc[1][vr]), fmaxf(sc[2][vr], sc[3][vr]));
#pragma unroll
            for (int off = 8; off > 0; off >>= 1)
                tm = fmaxf(tm, __shfl_xor(tm, off, 32));
            const float nm = fmaxf(mrow[vr], tm);
            newm[vr]  = nm;
            alpha[vr] = __expf(mrow[vr] - nm);
            mrow[vr]  = nm;
            rsum[vr]  = 0.0f;
        }
#pragma unroll
        for (int t = 0; t < 4; t++) {
#pragma unroll
            for (int vr = 0; vr < 8; vr++) {
                const float p = __expf(sc[t][vr] - newm[vr]);
                rsum[vr] += p;
                sPw[(vr + 8 * hf) * LDB + 16 * t + nn] = bf16_rne(p);
            }
        }
#pragma unroll
        for (int vr = 0; vr < 8; vr++) {
            float rs = rsum[vr];
#pragma unroll
            for (int off = 8; off > 0; off >>= 1)
                rs += __shfl_xor(rs, off, 32);
            lrow[vr] = lrow[vr] * alpha[vr] + rs;
#pragma unroll
            for (int t = 0; t < 4; t++) oacc[t][vr] *= alpha[vr];
        }

        // ---- O += P @ Vtile   (A = P via LDS round-trip, B^T rows = V^T)
#pragma unroll
        for (int c = 0; c < 2; c++) {
            const v16bf pf = load_fragA(sPw + 32 * c, LDB);
#pragma unroll
            for (int t = 0; t < 4; t++)
                oacc[t] = wmma_bf16(pf, load_fragB(sVt + 16 * t * LDB + 32 * c, LDB), oacc[t]);
        }
    }

    // ---- normalize and write O as bf16 in [B, S, H] layout
    const int b = bh >> 4, h = bh & (NH - 1);
#pragma unroll
    for (int vr = 0; vr < 8; vr++) {
        const float inv = 1.0f / lrow[vr];
        const int srow = q0 + 16 * wv + vr + 8 * hf;
#pragma unroll
        for (int t = 0; t < 4; t++) {
            const float val = oacc[t][vr] * inv;
            Obuf[((size_t)(b * SEQ + srow)) * HID + h * HD + 16 * t + nn] = bf16_rne(val);
        }
    }
}

// ---------------------------------------------------------------------------
// Kernel 3: output projection.  out = Obf16 @ Wo^T + bo  (fp32 out)
// Block tile 128x64, each wave 32x64 -> 8 WMMA / K-step
// grid = (NROWS/128, HID/64), block = 128
// ---------------------------------------------------------------------------
__global__ __launch_bounds__(128) void out_proj_kernel(
    const unsigned short* __restrict__ Obuf, const float* __restrict__ Wo,
    const float* __restrict__ bo, float* __restrict__ out)
{
    __shared__ __align__(16) unsigned short sA[128 * LDA];
    __shared__ __align__(16) unsigned short sW[64 * LDA];

    const int tid = threadIdx.x, wv = tid >> 5;
    const int lane = tid & 31, hf = lane >> 4, nn = lane & 15;
    const int n0 = blockIdx.x * 128, o0 = blockIdx.y * 64;

    v8f acc[2][4] = {};

    for (int k0 = 0; k0 < HID; k0 += 32) {
        __syncthreads();
        // A tile: already bf16, straight 16B copies (128 rows x 4 uint4)
#pragma unroll
        for (int i = 0; i < 4; i++) {
            const int li = tid + 128 * i;
            const int row = li >> 2, j = li & 3;
            *(uint4*)&sA[row * LDA + 8 * j] =
                *(const uint4*)(Obuf + (size_t)(n0 + row) * HID + k0 + 8 * j);
        }
        // W tile: fp32 -> bf16
#pragma unroll
        for (int i = 0; i < 4; i++) {
            const int li = tid + 128 * i;
            const int row = li >> 3, c4 = li & 7;
            const float4 fw = *(const float4*)(Wo + (size_t)(o0 + row) * HID + k0 + 4 * c4);
            uint2 pw;
            pw.x = pack2_bf16(fw.x, fw.y);
            pw.y = pack2_bf16(fw.z, fw.w);
            *(uint2*)&sW[row * LDA + 4 * c4] = pw;
            if (k0 + 32 < HID)
                __builtin_prefetch(Wo + (size_t)(o0 + row) * HID + k0 + 32 + 4 * c4, 0, 0);
        }
        __syncthreads();

        v16bf bfr[4];
#pragma unroll
        for (int t = 0; t < 4; t++)
            bfr[t] = load_fragB(sW + 16 * t * LDA, LDA);
        v16bf af0 = load_fragA(sA + (32 * wv) * LDA,      LDA);
        v16bf af1 = load_fragA(sA + (32 * wv + 16) * LDA, LDA);
#pragma unroll
        for (int t = 0; t < 4; t++) {
            acc[0][t] = wmma_bf16(af0, bfr[t], acc[0][t]);
            acc[1][t] = wmma_bf16(af1, bfr[t], acc[1][t]);
        }
    }

#pragma unroll
    for (int a = 0; a < 2; a++) {
#pragma unroll
        for (int t = 0; t < 4; t++) {
#pragma unroll
            for (int vr = 0; vr < 8; vr++) {
                const int n = n0 + 32 * wv + 16 * a + vr + 8 * hf;
                const int o = o0 + 16 * t + nn;
                out[(size_t)n * HID + o] = acc[a][t][vr] + bo[o];
            }
        }
    }
}

// ---------------------------------------------------------------------------
extern "C" void kernel_launch(void* const* d_in, const int* in_sizes, int n_in,
                              void* d_out, int out_size, void* d_ws, size_t ws_size,
                              hipStream_t stream) {
    const float* q  = (const float*)d_in[0];
    const float* k  = (const float*)d_in[1];
    const float* v  = (const float*)d_in[2];
    const float* Wq = (const float*)d_in[3];
    const float* bq = (const float*)d_in[4];
    const float* Wk = (const float*)d_in[5];
    const float* bk = (const float*)d_in[6];
    const float* Wv = (const float*)d_in[7];
    const float* bv = (const float*)d_in[8];
    const float* Wo = (const float*)d_in[9];
    const float* bo = (const float*)d_in[10];

    const size_t elems = (size_t)NROWS * HID;      // 4M bf16 per buffer
    unsigned short* Qp = (unsigned short*)d_ws;
    unsigned short* Kp = Qp + elems;
    unsigned short* Vp = Kp + elems;
    unsigned short* Ob = Vp + elems;               // total 32 MB workspace

    qkv_proj_kernel<<<dim3(NROWS / 128, HID / 64, 3), 128, 0, stream>>>(
        q, k, v, Wq, bq, Wk, bk, Wv, bv, Qp, Kp, Vp);
    attn_kernel<<<dim3(SEQ / 64, NB * NH), 128, 0, stream>>>(Qp, Kp, Vp, Ob);
    out_proj_kernel<<<dim3(NROWS / 128, HID / 64), 128, 0, stream>>>(
        Ob, Wo, bo, (float*)d_out);
}
